// ROBERTA_FOL_R_41266045780249
// MI455X (gfx1250) — matrix operand order repaired and checked by reading, not compiled, not run on hardware
//
#include <hip/hip_runtime.h>

// ---------------------------------------------------------------------------
// Types for CDNA5 WMMA
// ---------------------------------------------------------------------------
typedef __attribute__((ext_vector_type(16))) __bf16 v16bf;
typedef __attribute__((ext_vector_type(8)))  float  v8f;

#define Bb 64
#define Ss 256
#define Hh 768
#define RR (255 * 64)   // 16320 batched rows

// ---------------------------------------------------------------------------
// bf16 helpers (round-to-nearest-even)
// ---------------------------------------------------------------------------
__device__ inline unsigned short f2bf(float f) {
    union { float f; unsigned int u; } x; x.f = f;
    unsigned int u = x.u + 0x7FFFu + ((x.u >> 16) & 1u);
    return (unsigned short)(u >> 16);
}
__device__ inline float bf2f(unsigned short s) {
    union { unsigned int u; float f; } x; x.u = ((unsigned int)s) << 16;
    return x.f;
}

// ---------------------------------------------------------------------------
// CDNA5 async Global->LDS copy (16B per lane) + asynccnt wait.
// global_load_async_to_lds_b128: VDST = LDS byte address (VGPR),
// VADDR = 64-bit global address.  Tracked by ASYNCcnt.
// ---------------------------------------------------------------------------
__device__ inline void async_cp16(unsigned lds_byte, const void* g) {
    asm volatile("global_load_async_to_lds_b128 %0, %1, off"
                 :: "v"(lds_byte),
                    "v"((unsigned long long)(size_t)g)
                 : "memory");
}
__device__ inline void wait_async0() {
    asm volatile("s_wait_asynccnt 0" ::: "memory");
}

// ---------------------------------------------------------------------------
// Fragment loaders matching CDNA5 WMMA VGPR layouts (ISA 7.12.2).
// A (16x32 bf16, row-major source, row stride K):
//   lane m=lane&15, half=lane>>4; elems 0..7 -> K = k0 + 8*half + e (contig),
//   elems 8..15 -> K = k0 + 16 + 8*half + (e-8) (contig).
// B (32x16 bf16) from N-major (transposed) weights Wt[n][k]:
//   lane n=lane&15; elems 0..7 -> K = k0 + 8*half + e, elems 8..15 -> K = k0 + 16 + 8*half + (e-8).
// ---------------------------------------------------------------------------
__device__ inline v16bf load_a_frag(const unsigned short* A, int row, int K,
                                    int k0, int half) {
    union { v16bf v; uint4 q[2]; } f;
    const unsigned short* p = A + (size_t)row * K + k0 + half * 8;
    f.q[0] = *(const uint4*)(p);
    f.q[1] = *(const uint4*)(p + 16);
    return f.v;
}
__device__ inline v16bf load_b_frag(const unsigned short* Bt, int col, int K,
                                    int k0, int half) {
    union { v16bf v; uint4 q[2]; } f;
    const unsigned short* p = Bt + (size_t)col * K + k0 + half * 8;
    f.q[0] = *(const uint4*)(p);
    f.q[1] = *(const uint4*)(p + 16);
    return f.v;
}
__device__ inline v8f wmma_bf16(v16bf a, v16bf b, v8f c) {
    return __builtin_amdgcn_wmma_f32_16x16x32_bf16(false, a, false, b,
                                                   (short)0, c, false, false);
}

// ---------------------------------------------------------------------------
// Kernel: transpose fp32 weight (K x N row-major, starting at rowOff) into
// bf16 N-major Wt[n][k].
// ---------------------------------------------------------------------------
__global__ void k_transpose_bf(const float* __restrict__ W,
                               unsigned short* __restrict__ Wt,
                               int K, int N, int rowOff) {
    long total = (long)K * N;
    for (long idx = blockIdx.x * (long)blockDim.x + threadIdx.x; idx < total;
         idx += (long)gridDim.x * blockDim.x) {
        int n = (int)(idx / K);
        int k = (int)(idx % K);
        Wt[idx] = f2bf(W[(size_t)(rowOff + k) * N + n]);
    }
}

// ---------------------------------------------------------------------------
// Kernel: build bf16 X (t-major) and leaky(X) for t = 1..255.
// Row r = (t-1)*64 + b. hs layout (B,S,H).
// ---------------------------------------------------------------------------
__global__ void k_xprep(const float* __restrict__ hs,
                        unsigned short* __restrict__ Xb,
                        unsigned short* __restrict__ Lb) {
    long total = (long)RR * Hh;
    for (long idx = blockIdx.x * (long)blockDim.x + threadIdx.x; idx < total;
         idx += (long)gridDim.x * blockDim.x) {
        int r = (int)(idx / Hh);
        int h = (int)(idx % Hh);
        int t = r / Bb + 1;
        int b = r % Bb;
        float x = hs[((size_t)b * Ss + t) * Hh + h];
        Xb[idx] = f2bf(x);
        Lb[idx] = f2bf(x >= 0.f ? x : -0.2f * x);
    }
}

// ---------------------------------------------------------------------------
// Kernel: per-row dot with w_and (+ b_and). One wave per row (wave32).
// mode 0: row r -> hs[(b*S + t)*H] with t=r/64+1, b=r%64 ; mode 1: X + r*H.
// ---------------------------------------------------------------------------
__global__ void k_rowdot(const float* __restrict__ X,
                         const float* __restrict__ w,
                         const float* __restrict__ bscal,
                         float* __restrict__ out, int R, int mode) {
    int gw = blockIdx.x * (blockDim.x >> 5) + (threadIdx.x >> 5);
    int lane = threadIdx.x & 31;
    if (gw >= R) return;
    const float* base;
    if (mode == 0) {
        int t = gw / Bb + 1, b = gw % Bb;
        base = X + ((size_t)b * Ss + t) * Hh;
    } else {
        base = X + (size_t)gw * Hh;
    }
    float p = 0.f;
    for (int j = lane; j < Hh; j += 32) p += base[j] * w[j];
    p += __shfl_xor(p, 16);
    p += __shfl_xor(p, 8);
    p += __shfl_xor(p, 4);
    p += __shfl_xor(p, 2);
    p += __shfl_xor(p, 1);
    if (lane == 0) out[gw] = p + bscal[0];
}

// ---------------------------------------------------------------------------
// Generic bf16 WMMA GEMM:  C[M,N] = A[M,K] * Bt[N,K]^T + bias[N]
// Block 256 threads = 8 waves (4x2); block tile 128x128, BK=32.
// Double-buffered LDS tiles filled by async Global->LDS DMA
// (global_load_async_to_lds_b128), WMMA consumes from LDS (ds_load_b128).
// The ASYNCcnt wait is AFTER compute so DMA of tile k+1 overlaps WMMA of k.
// ---------------------------------------------------------------------------
__global__ __launch_bounds__(256) void k_gemm_bf16(
    const unsigned short* __restrict__ A, const unsigned short* __restrict__ Bt,
    const float* __restrict__ bias, float* __restrict__ C,
    int M, int N, int K) {
    __shared__ unsigned short ldsA[2][128 * 32];   // 8KB per buffer
    __shared__ unsigned short ldsB[2][128 * 32];   // 8KB per buffer

    const int tid  = threadIdx.x;
    const int lane = tid & 31;
    const int wave = tid >> 5;
    const int half = lane >> 4;
    const int l15  = lane & 15;
    const int mw = wave >> 1;                 // 0..3
    const int nw = wave & 1;                  // 0..1
    const int m0b = blockIdx.x * 128;
    const int n0b = blockIdx.y * 128;

    // staging assignment: thread copies two 16B chunks of A and two of B
    const int rowc = tid >> 2;                // 0..63
    const int part = tid & 3;                 // 16B chunk within a 64B row
    const int ar0 = min(m0b + rowc,      M - 1);
    const int ar1 = min(m0b + rowc + 64, M - 1);
    const int br0 = min(n0b + rowc,      N - 1);
    const int br1 = min(n0b + rowc + 64, N - 1);

    auto stage = [&](int buf, int k0) {
        const unsigned short* pa0 = A + (size_t)ar0 * K + k0 + part * 8;
        const unsigned short* pa1 = A + (size_t)ar1 * K + k0 + part * 8;
        const unsigned short* pb0 = Bt + (size_t)br0 * K + k0 + part * 8;
        const unsigned short* pb1 = Bt + (size_t)br1 * K + k0 + part * 8;
        async_cp16((unsigned)(size_t)&ldsA[buf][rowc * 32 + part * 8], pa0);
        async_cp16((unsigned)(size_t)&ldsA[buf][(rowc + 64) * 32 + part * 8], pa1);
        async_cp16((unsigned)(size_t)&ldsB[buf][rowc * 32 + part * 8], pb0);
        async_cp16((unsigned)(size_t)&ldsB[buf][(rowc + 64) * 32 + part * 8], pb1);
    };

    v8f acc[2][4];
#pragma unroll
    for (int i = 0; i < 2; ++i)
#pragma unroll
        for (int j = 0; j < 4; ++j)
#pragma unroll
            for (int e = 0; e < 8; ++e) acc[i][j][e] = 0.f;

    stage(0, 0);
    wait_async0();
    __syncthreads();

    const int nk = K >> 5;
    for (int kk = 0; kk < nk; ++kk) {
        const int buf = kk & 1;
        if (kk + 1 < nk) stage(buf ^ 1, (kk + 1) << 5);   // async prefetch

        const unsigned short* sA = &ldsA[buf][0];
        const unsigned short* sB = &ldsB[buf][0];
        v16bf a0 = load_a_frag(sA, mw * 32 + l15, 32, 0, half);
        v16bf a1 = load_a_frag(sA, mw * 32 + 16 + l15, 32, 0, half);
#pragma unroll
        for (int j = 0; j < 4; ++j) {
            v16bf bf = load_b_frag(sB, nw * 64 + j * 16 + l15, 32, 0, half);
            acc[0][j] = wmma_bf16(a0, bf, acc[0][j]);
            acc[1][j] = wmma_bf16(a1, bf, acc[1][j]);
        }
        wait_async0();      // next tile landed (overlapped with WMMA above)
        __syncthreads();
    }

    const int m0 = m0b + mw * 32;
    const int n0 = n0b + nw * 64;
#pragma unroll
    for (int i = 0; i < 2; ++i)
#pragma unroll
        for (int j = 0; j < 4; ++j) {
            int col = n0 + j * 16 + l15;
            if (col >= N) continue;
            float bj = bias ? bias[col] : 0.f;
#pragma unroll
            for (int r = 0; r < 8; ++r) {
                int row = m0 + i * 16 + r + half * 8;
                if (row < M) C[(size_t)row * N + col] = acc[i][j][r] + bj;
            }
        }
}

// ---------------------------------------------------------------------------
// Sequential FOL scan: single persistent workgroup (1024 thr = 32 waves).
// acc (64x768) lives in LDS as bf16.  Per step t:
//   phase 1: sacc[b] = acc . w_and + b_and (wave reductions) and
//            R = acc @ WtTop via WMMA (192 tiles, 6 per wave, B from L2)
//   phase 2: op-select combine with precomputed Uo / Yn / sx / sn
//   phase 3: write new acc (bf16) back to LDS
// Epilogue writes cat = [acc, 0.2*pooled] in bf16 for the final GEMM.
// ---------------------------------------------------------------------------
__global__ __launch_bounds__(1024) void k_fol_scan(
    const float* __restrict__ hs, const int* __restrict__ indices,
    const float* __restrict__ Yn, const float* __restrict__ Uo,
    const float* __restrict__ sx, const float* __restrict__ sn,
    const unsigned short* __restrict__ WtTop,
    const float* __restrict__ w_and, const float* __restrict__ b_and,
    const float* __restrict__ pooled, unsigned short* __restrict__ cat_bf) {
    extern __shared__ unsigned char smem[];
    unsigned short* accb = (unsigned short*)smem;              // 64*768 bf16
    float* sacc = (float*)(smem + (size_t)Bb * Hh * 2);        // 64 f32

    const int tid  = threadIdx.x;
    const int lane = tid & 31;
    const int wave = tid >> 5;
    const int half = lane >> 4;
    const int l15  = lane & 15;
    const int m_tile = wave >> 3;   // 0..3  (16 rows each)
    const int nset   = wave & 7;    // 0..7  (6 n-tiles each)

    // init acc = hidden_state[:, 0, :]
    for (int i = tid; i < Bb * Hh; i += 1024) {
        int b = i / Hh, h = i % Hh;
        accb[i] = f2bf(hs[((size_t)b * Ss) * Hh + h]);
    }
    __syncthreads();

    const float band = b_and[0];
    v8f acc6[6];

    for (int t = 1; t < Ss; ++t) {
        // ---- phase 1a: sacc reductions (2 rows per wave) ----
#pragma unroll
        for (int rr = 0; rr < 2; ++rr) {
            int row = wave * 2 + rr;
            float p = 0.f;
            for (int j = lane; j < Hh; j += 32)
                p += bf2f(accb[row * Hh + j]) * w_and[j];
            p += __shfl_xor(p, 16);
            p += __shfl_xor(p, 8);
            p += __shfl_xor(p, 4);
            p += __shfl_xor(p, 2);
            p += __shfl_xor(p, 1);
            if (lane == 0) sacc[row] = p + band;
        }
        // ---- phase 1b: WMMA  acc @ WtTop  (A from LDS, B from L2) ----
#pragma unroll
        for (int j = 0; j < 6; ++j)
#pragma unroll
            for (int e = 0; e < 8; ++e) acc6[j][e] = 0.f;
        for (int k0 = 0; k0 < Hh; k0 += 32) {
            v16bf a = load_a_frag(accb, m_tile * 16 + l15, Hh, k0, half);
#pragma unroll
            for (int j = 0; j < 6; ++j) {
                int n = (nset * 6 + j) * 16 + l15;
                v16bf bf = load_b_frag(WtTop, n, Hh, k0, half);
                acc6[j] = wmma_bf16(a, bf, acc6[j]);
            }
        }
        __syncthreads();   // sacc visible; all reads of accb done

        // ---- phase 2: op-select combine (results kept in acc6) ----
        const int rbase = (t - 1) * Bb;
#pragma unroll
        for (int r = 0; r < 8; ++r) {
            int b   = m_tile * 16 + r + half * 8;
            int op  = indices[(size_t)b * Ss + t] % 3;
            float sa  = sacc[b];
            int ridx  = rbase + b;
            float sxv = sx[ridx];
            float snv = sn[ridx];
#pragma unroll
            for (int j = 0; j < 6; ++j) {
                int h = (nset * 6 + j) * 16 + l15;
                float oldv = bf2f(accb[b * Hh + h]);
                float v;
                if (op == 0)
                    v = oldv * sa + hs[((size_t)b * Ss + t) * Hh + h] * sxv;
                else if (op == 1)
                    v = acc6[j][r] + Uo[(size_t)ridx * Hh + h];
                else
                    v = oldv * sa + Yn[(size_t)ridx * Hh + h] * snv;
                acc6[j][r] = v;
            }
        }
        __syncthreads();   // everyone finished reading old acc

        // ---- phase 3: write new acc ----
#pragma unroll
        for (int r = 0; r < 8; ++r) {
            int b = m_tile * 16 + r + half * 8;
#pragma unroll
            for (int j = 0; j < 6; ++j) {
                int h = (nset * 6 + j) * 16 + l15;
                accb[b * Hh + h] = f2bf(acc6[j][r]);
            }
        }
        __syncthreads();
    }

    // ---- epilogue: cat = [fol_out, 0.2*pooled] (bf16) ----
#pragma unroll
    for (int r = 0; r < 8; ++r) {
        int b = m_tile * 16 + r + half * 8;
#pragma unroll
        for (int j = 0; j < 6; ++j) {
            int h = (nset * 6 + j) * 16 + l15;
            cat_bf[(size_t)b * 1536 + h]       = f2bf(acc6[j][r]);
            cat_bf[(size_t)b * 1536 + Hh + h]  =
                f2bf(0.2f * pooled[(size_t)b * Hh + h]);
        }
    }
}

// ---------------------------------------------------------------------------
// logits[g,b,k] = out[b] . word_emb[pidx[g,k]]   (one wave per output)
// ---------------------------------------------------------------------------
__global__ void k_logits(const float* __restrict__ outD,
                         const float* __restrict__ emb,
                         const int* __restrict__ pidx,
                         float* __restrict__ out) {
    int gw = blockIdx.x * (blockDim.x >> 5) + (threadIdx.x >> 5);
    int lane = threadIdx.x & 31;
    if (gw >= 3 * Bb * 4) return;
    int k = gw & 3;
    int b = (gw >> 2) & 63;
    int g = gw >> 8;
    int row = pidx[g * 4 + k];
    const float* e = emb + (size_t)row * Hh;
    const float* o = outD + (size_t)b * Hh;
    float p = 0.f;
    for (int j = lane; j < Hh; j += 32) p += o[j] * e[j];
    p += __shfl_xor(p, 16);
    p += __shfl_xor(p, 8);
    p += __shfl_xor(p, 4);
    p += __shfl_xor(p, 2);
    p += __shfl_xor(p, 1);
    if (lane == 0) out[gw] = p;
}

// ---------------------------------------------------------------------------
// Host-side orchestration
// ---------------------------------------------------------------------------
extern "C" void kernel_launch(void* const* d_in, const int* in_sizes, int n_in,
                              void* d_out, int out_size, void* d_ws,
                              size_t ws_size, hipStream_t stream) {
    const int*   indices = (const int*)d_in[0];
    const float* hs      = (const float*)d_in[1];
    const float* pooled  = (const float*)d_in[2];
    const float* W_not   = (const float*)d_in[3];
    const float* b_not   = (const float*)d_in[4];
    const float* W_and   = (const float*)d_in[5];
    const float* b_and   = (const float*)d_in[6];
    const float* W_or    = (const float*)d_in[7];
    const float* b_or    = (const float*)d_in[8];
    const float* W_d1    = (const float*)d_in[9];
    const float* b_d1    = (const float*)d_in[10];
    const float* wemb    = (const float*)d_in[11];
    const int*   pidx    = (const int*)d_in[12];
    float* out = (float*)d_out;

    unsigned char* ws = (unsigned char*)d_ws;
    size_t o = 0;
    unsigned short* Wtn  = (unsigned short*)(ws + o); o += (size_t)Hh * Hh * 2;
    unsigned short* Wtop = (unsigned short*)(ws + o); o += (size_t)Hh * Hh * 2;
    unsigned short* Wbot = (unsigned short*)(ws + o); o += (size_t)Hh * Hh * 2;
    unsigned short* Wtd1 = (unsigned short*)(ws + o); o += (size_t)Hh * 1536 * 2;
    unsigned short* Xb   = (unsigned short*)(ws + o); o += (size_t)RR * Hh * 2;
    unsigned short* Lb   = (unsigned short*)(ws + o); o += (size_t)RR * Hh * 2;
    float* Yn = (float*)(ws + o); o += (size_t)RR * Hh * 4;
    float* Uo = (float*)(ws + o); o += (size_t)RR * Hh * 4;
    float* sx = (float*)(ws + o); o += (size_t)RR * 4 + 128;
    float* sn = (float*)(ws + o); o += (size_t)RR * 4 + 128;
    unsigned short* cat = (unsigned short*)(ws + o); o += (size_t)Bb * 1536 * 2;
    float* outD = (float*)(ws + o); o += (size_t)Bb * Hh * 4;
    (void)ws_size; (void)in_sizes; (void)n_in; (void)out_size;

    // 1) weight transposes (fp32 -> bf16 N-major)
    k_transpose_bf<<<2304, 256, 0, stream>>>(W_not, Wtn, Hh, Hh, 0);
    k_transpose_bf<<<2304, 256, 0, stream>>>(W_or, Wtop, Hh, Hh, 0);
    k_transpose_bf<<<2304, 256, 0, stream>>>(W_or, Wbot, Hh, Hh, Hh);
    k_transpose_bf<<<4608, 256, 0, stream>>>(W_d1, Wtd1, 1536, Hh, 0);

    // 2) X / leaky(X) in bf16, t-major
    k_xprep<<<8192, 256, 0, stream>>>(hs, Xb, Lb);

    // 3) sx = x_t . w_and + b_and
    k_rowdot<<<2040, 256, 0, stream>>>(hs, W_and, b_and, sx, RR, 0);

    // 4) batched GEMMs (all t at once):
    //    Yn = leaky(X) @ W_not + b_not ;  Uo = X @ W_or[H:] + b_or
    dim3 g1(128, 6);
    k_gemm_bf16<<<g1, 256, 0, stream>>>(Lb, Wtn,  b_not, Yn, RR, Hh, Hh);
    k_gemm_bf16<<<g1, 256, 0, stream>>>(Xb, Wbot, b_or,  Uo, RR, Hh, Hh);

    // 5) sn = notnet(x_t) . w_and + b_and
    k_rowdot<<<2040, 256, 0, stream>>>(Yn, W_and, b_and, sn, RR, 1);

    // 6) sequential scan (persistent single workgroup, LDS-resident acc)
    size_t smem = (size_t)Bb * Hh * 2 + 64 * 4;
    k_fol_scan<<<1, 1024, smem, stream>>>(hs, indices, Yn, Uo, sx, sn, Wtop,
                                          W_and, b_and, pooled, cat);

    // 7) out = cat @ W_d1 + b_d1  (M=64, K=1536, N=768)
    dim3 g2(1, 6);
    k_gemm_bf16<<<g2, 256, 0, stream>>>(cat, Wtd1, b_d1, outD, Bb, Hh, 1536);

    // 8) logits
    k_logits<<<96, 256, 0, stream>>>(outD, wemb, pidx, out);
}
